// ChildHAggregation_3925600108794
// MI455X (gfx1250) — matrix operand order repaired and checked by reading, not compiled, not run on hardware
//
#include <hip/hip_runtime.h>
#include <cmath>

// ---------------------------------------------------------------------------
// MI455X (gfx1250) fused ChildHAggregation forward, v2.
// Compute-bound (~275 GFLOP vs ~400MB HBM traffic) -> all GEMMs on
// v_wmma_f32_16x16x32_f16 (f32 accumulate).
// v2: 32 rows per workgroup. Every B (weight) fragment loaded from L2 is
// reused across 2-4 M-tiles in registers -> halves L2 weight traffic
// (7.3MB weight set re-read 1024x instead of 2048x) and raises the
// wmma : vmem ratio to 2-4 : 1 per fragment.
// Activations staged once in LDS as f16 (~193 KB of the 320 KB WGP budget).
// ---------------------------------------------------------------------------

typedef __attribute__((ext_vector_type(8)))  _Float16 v8h;
typedef __attribute__((ext_vector_type(16))) _Float16 v16h;
typedef __attribute__((ext_vector_type(8)))  float    v8f;

#define HALF_D 512
#define DIM_D  1024
#define ROWS   32     // batch rows per workgroup
#define NTHREADS 256  // 8 wave32

// ---------------------------------------------------------------------------
// Weight conversion: f32 row-major [K][N] -> f16 fragments.
// Fragment = one 32(K) x 16(N) tile, 512 f16 contiguous:
//   lane l (0..31): n = nt*16 + (l&15); elements j=0..15: k = kt*32 + (l>>4)*8 + (j&7) + (j>>3)*16
//   index within fragment = l*16 + j  (each lane loads 32B contiguous)
// Tiles ordered: frag_id = nt * (K/32) + kt.
// ---------------------------------------------------------------------------
__global__ void convert_weight_f16(const float* __restrict__ src,
                                   _Float16* __restrict__ dst,
                                   int K, int N) {
    int idx = blockIdx.x * NTHREADS + threadIdx.x;
    int total = K * N;
    if (idx >= total) return;
    int frag   = idx >> 9;
    int within = idx & 511;
    int lane = within >> 4;
    int j    = within & 15;
    int ktiles = K >> 5;
    int nt = frag / ktiles;
    int kt = frag - nt * ktiles;
    int k = kt * 32 + (lane >> 4) * 8 + (j & 7) + (j >> 3) * 16;
    int n = nt * 16 + (lane & 15);
    dst[idx] = (_Float16)src[(size_t)k * N + n];
}

// ---------------------------------------------------------------------------
// Fragment loaders
// ---------------------------------------------------------------------------
__device__ inline v16h make16(v8h lo, v8h hi) {
    return __builtin_shufflevector(lo, hi, 0,1,2,3,4,5,6,7,8,9,10,11,12,13,14,15);
}

// A fragment (16x32 f16) from LDS row-major [M][ld]:
//   lanes 0-15: m=lane, K {kb..kb+7, kb+16..kb+23}
//   lanes 16-31: m=lane-16, K {kb+8..kb+15, kb+24..kb+31}
__device__ inline v16h ldA(const _Float16* base, int ld, int kbase) {
    int lane = threadIdx.x & 31;
    int m  = lane & 15;
    int ks = (lane >> 4) * 8;
    const _Float16* p = base + m * ld + kbase + ks;
    return make16(*(const v8h*)p, *(const v8h*)(p + 16));   // 2x ds_load_b128
}

// A fragment for the token-stacked hlr matrix: token-row (mt*16+m) -> batch
// mt*8+(m>>1), token (m&1) selecting hl/hr half of sAct[batch][0..1023].
__device__ inline v16h ldA_tok(const _Float16* sact, int mt, int kbase) {
    int lane = threadIdx.x & 31;
    int m  = lane & 15;
    int ks = (lane >> 4) * 8;
    const _Float16* p = sact + (mt * 8 + (m >> 1)) * DIM_D + (m & 1) * HALF_D + kbase + ks;
    return make16(*(const v8h*)p, *(const v8h*)(p + 16));
}

// B fragment (32x16 f16) from pre-swizzled global weight buffer (32B/lane).
__device__ inline v16h ldB(const _Float16* frag, int kt, int nt, int ktiles) {
    int lane = threadIdx.x & 31;
    return *(const v16h*)(frag + (((size_t)nt * ktiles + kt) << 9) + lane * 16);
}

__device__ inline v8f wmma16(v16h a, v16h b, v8f c) {
    return __builtin_amdgcn_wmma_f32_16x16x32_f16(false, a, false, b, (short)0, c,
                                                  false, false);
}

// ---------------------------------------------------------------------------
// Fused forward kernel
// ---------------------------------------------------------------------------
__global__ __launch_bounds__(NTHREADS)
void child_h_agg_fused(
    const float* __restrict__ hl, const float* __restrict__ hr,
    const float* __restrict__ xw, const float* __restrict__ xh,
    const _Float16* __restrict__ wqU,  const float* __restrict__ qUb,
    const _Float16* __restrict__ wqWu, const float* __restrict__ qWub,
    const _Float16* __restrict__ wqWb, const float* __restrict__ qWbb,
    const _Float16* __restrict__ wkU,  const float* __restrict__ kUb,
    const _Float16* __restrict__ wkWu, const float* __restrict__ kWub,
    const _Float16* __restrict__ wkWb, const float* __restrict__ kWbb,
    const float* __restrict__ alpha,   const float* __restrict__ beta,
    const _Float16* __restrict__ whU,  const float* __restrict__ hUb,
    const _Float16* __restrict__ whWu, const float* __restrict__ hWub,
    const _Float16* __restrict__ whWb, const float* __restrict__ hWbb,
    const _Float16* __restrict__ wlU,  const float* __restrict__ lUb,
    const _Float16* __restrict__ wlWu, const float* __restrict__ lWub,
    const _Float16* __restrict__ wlWb, const float* __restrict__ lWbb,
    float* __restrict__ out)
{
    __shared__ _Float16 sAct[ROWS][DIM_D];       // hl|hr, later normalized x (64KB)
    __shared__ _Float16 sXH [ROWS][HALF_D];      // 32KB
    __shared__ _Float16 sXW [ROWS][DIM_D];       // 64KB
    __shared__ _Float16 sGate[8][4][ROWS][16];   // per-wave gate tiles, f16 (32KB)
    __shared__ float    sScore[ROWS][4];         // q.k^T partials (2x2 per row)

    const int tid  = threadIdx.x;
    const int wave = tid >> 5;
    const int lane = tid & 31;
    const int hi8  = (lane >> 4) * 8;    // C/D fragment M offset
    const int nl   = lane & 15;          // C/D fragment N (local)
    const long rowBase = (long)blockIdx.x * ROWS;

    // ---- Phase 0: stage activations to LDS as f16 --------------------------
    for (int i = tid; i < ROWS * HALF_D; i += NTHREADS) {
        int r = i >> 9, c = i & (HALF_D - 1);
        long g = (rowBase + r) * HALF_D + c;
        sAct[r][c]          = (_Float16)hl[g];
        sAct[r][HALF_D + c] = (_Float16)hr[g];
        sXH[r][c]           = (_Float16)xh[g];
    }
    for (int i = tid; i < ROWS * DIM_D; i += NTHREADS) {
        int r = i >> 10, c = i & (DIM_D - 1);
        sXW[r][c] = (_Float16)xw[(rowBase + r) * DIM_D + c];
    }
    if (tid < ROWS * 4) (&sScore[0][0])[tid] = 0.0f;
    __syncthreads();

    // ---- Phase 1: q/k gated GEMMs, fused into 2x2 score dot-products -------
    // Each wave owns 4 N-tiles of the d=512 reduction dimension.
    for (int t = 0; t < 4; ++t) {
        int nt = wave * 4 + t;
        int ng = nt * 16 + nl;
        float qub = qUb[ng], kub = kUb[ng];
        float qwub = qWub[ng], qwbb = qWbb[ng], kwub = kWub[ng], kwbb = kWbb[ng];

        // gate GEMMs: 2 M-tiles (32 batch rows) share each B fragment
        {
            v8f gq[2], bq[2], gk[2], bk[2];
            #pragma unroll
            for (int g = 0; g < 2; ++g) { gq[g] = 0.0f; bq[g] = 0.0f; gk[g] = 0.0f; bk[g] = 0.0f; }
            for (int kt = 0; kt < 16; ++kt) {
                v16h b0 = ldB(wqWu, kt, nt, 16);
                v16h b1 = ldB(wqWb, kt, nt, 16);
                v16h b2 = ldB(wkWu, kt, nt, 16);
                v16h b3 = ldB(wkWb, kt, nt, 16);
                #pragma unroll
                for (int g = 0; g < 2; ++g) {
                    v16h a = ldA(&sXH[g * 16][0], HALF_D, kt * 32);
                    gq[g] = wmma16(a, b0, gq[g]);
                    bq[g] = wmma16(a, b1, bq[g]);
                    gk[g] = wmma16(a, b2, gk[g]);
                    bk[g] = wmma16(a, b3, bk[g]);
                }
            }
            #pragma unroll
            for (int g = 0; g < 2; ++g) {
                #pragma unroll
                for (int v = 0; v < 8; ++v) {
                    int m = g * 16 + hi8 + v;
                    sGate[wave][0][m][nl] = (_Float16)(gq[g][v] + qwub);
                    sGate[wave][1][m][nl] = (_Float16)(bq[g][v] + qwbb);
                    sGate[wave][2][m][nl] = (_Float16)(gk[g][v] + kwub);
                    sGate[wave][3][m][nl] = (_Float16)(bk[g][v] + kwbb);
                }
            }
        }

        // u GEMMs: 4 token M-tiles (64 token rows) share each B fragment
        v8f uq[4], uk[4];
        #pragma unroll
        for (int mt = 0; mt < 4; ++mt) { uq[mt] = 0.0f; uk[mt] = 0.0f; }
        for (int kt = 0; kt < 16; ++kt) {
            v16h bu = ldB(wqU, kt, nt, 16);
            v16h bv = ldB(wkU, kt, nt, 16);
            #pragma unroll
            for (int mt = 0; mt < 4; ++mt) {
                v16h a = ldA_tok(&sAct[0][0], mt, kt * 32);
                uq[mt] = wmma16(a, bu, uq[mt]);
                uk[mt] = wmma16(a, bv, uk[mt]);
            }
        }

        // epilogue + 2x2 score partial accumulation
        #pragma unroll
        for (int mt = 0; mt < 4; ++mt) {
            float qv[8], kv[8];
            #pragma unroll
            for (int v = 0; v < 8; ++v) {
                int gidx = mt * 8 + ((hi8 + v) >> 1);   // batch row 0..31
                float g0 = (float)sGate[wave][0][gidx][nl];
                float g1 = (float)sGate[wave][1][gidx][nl];
                float g2 = (float)sGate[wave][2][gidx][nl];
                float g3 = (float)sGate[wave][3][gidx][nl];
                qv[v] = (uq[mt][v] + qub) * g0 + g1;
                kv[v] = (uk[mt][v] + kub) * g2 + g3;
            }
            #pragma unroll
            for (int b = 0; b < 4; ++b) {
                #pragma unroll
                for (int ij = 0; ij < 4; ++ij) {
                    int i = ij >> 1, j = ij & 1;
                    float pr = qv[2 * b + i] * kv[2 * b + j];
                    pr += __shfl_xor(pr, 1, 16);
                    pr += __shfl_xor(pr, 2, 16);
                    pr += __shfl_xor(pr, 4, 16);
                    pr += __shfl_xor(pr, 8, 16);
                    if (nl == 0) {
                        int batch = mt * 8 + (hi8 >> 1) + b;
                        atomicAdd(&sScore[batch][ij], pr);   // ds_add_f32
                    }
                }
            }
        }
    }
    __syncthreads();

    // ---- Phase 2: softmax(2x2), attention mix, LayerNorm (ddof=1) ----------
    // two passes of 16 rows; each row by 16 lockstep lanes of one wave
    for (int rp = 0; rp < 2; ++rp) {
        const int r  = rp * 16 + (tid >> 4);
        const int cg = tid & 15;
        const float scale = 0.04419417382415922f;   // 1/sqrt(512)
        float s00 = sScore[r][0] * scale, s01 = sScore[r][1] * scale;
        float s10 = sScore[r][2] * scale, s11 = sScore[r][3] * scale;
        float m0 = fmaxf(s00, s01), m1 = fmaxf(s10, s11);
        float e00 = __expf(s00 - m0), e01 = __expf(s01 - m0);
        float e10 = __expf(s10 - m1), e11 = __expf(s11 - m1);
        float p00 = e00 / (e00 + e01), p01 = e01 / (e00 + e01);
        float p10 = e10 / (e10 + e11), p11 = e11 / (e10 + e11);

        float xv[64];
        float sum = 0.0f, sq = 0.0f;
        #pragma unroll
        for (int j = 0; j < 64; ++j) {
            int c  = cg * 64 + j;
            int i  = c >> 9;            // token index of output column
            int cc = c & (HALF_D - 1);
            float base = (float)sAct[r][c];
            float a0 = (float)sAct[r][cc];
            float a1 = (float)sAct[r][HALF_D + cc];
            float mix = (i == 0) ? (p00 * a0 + p01 * a1) : (p10 * a0 + p11 * a1);
            float val = base + mix;
            xv[j] = val; sum += val; sq += val * val;
        }
        #pragma unroll
        for (int off = 1; off < 16; off <<= 1) {
            sum += __shfl_xor(sum, off, 16);
            sq  += __shfl_xor(sq,  off, 16);
        }
        float mean = sum * (1.0f / 1024.0f);
        float var  = (sq - 1024.0f * mean * mean) * (1.0f / 1023.0f);
        float inv  = 1.0f / (sqrtf(fmaxf(var, 0.0f)) + 1e-6f);
        // lockstep 16-lane group: all reads above precede these writes
        #pragma unroll
        for (int j = 0; j < 64; ++j) {
            int c = cg * 64 + j;
            sAct[r][c] = (_Float16)(alpha[c] * (xv[j] - mean) * inv + beta[c]);
        }
    }
    __syncthreads();

    // ---- Phase 3: hidden + leaf gated GEMMs, epilogue, store ---------------
    for (int t = 0; t < 4; ++t) {
        int nt = wave * 4 + t;
        int ng = nt * 16 + nl;

        // main products: 2 M-tiles share each B fragment
        v8f ah[2], al[2];
        #pragma unroll
        for (int mt = 0; mt < 2; ++mt) { ah[mt] = 0.0f; al[mt] = 0.0f; }
        for (int kt = 0; kt < 32; ++kt) {
            v16h bU = ldB(whU, kt, nt, 32);
            v16h bL = ldB(wlU, kt, nt, 32);
            #pragma unroll
            for (int mt = 0; mt < 2; ++mt) {
                ah[mt] = wmma16(ldA(&sAct[mt * 16][0], DIM_D, kt * 32), bU, ah[mt]);
                al[mt] = wmma16(ldA(&sXW[mt * 16][0],  DIM_D, kt * 32), bL, al[mt]);
            }
        }
        // gates
        v8f gh[2], bh[2], gl[2], bl[2];
        #pragma unroll
        for (int mt = 0; mt < 2; ++mt) { gh[mt] = 0.0f; bh[mt] = 0.0f; gl[mt] = 0.0f; bl[mt] = 0.0f; }
        for (int kt = 0; kt < 16; ++kt) {
            v16h b0 = ldB(whWu, kt, nt, 16);
            v16h b1 = ldB(whWb, kt, nt, 16);
            v16h b2 = ldB(wlWu, kt, nt, 16);
            v16h b3 = ldB(wlWb, kt, nt, 16);
            #pragma unroll
            for (int mt = 0; mt < 2; ++mt) {
                v16h a = ldA(&sXH[mt * 16][0], HALF_D, kt * 32);
                gh[mt] = wmma16(a, b0, gh[mt]);
                bh[mt] = wmma16(a, b1, bh[mt]);
                gl[mt] = wmma16(a, b2, gl[mt]);
                bl[mt] = wmma16(a, b3, bl[mt]);
            }
        }
        float hub = hUb[ng], hwub = hWub[ng], hwbb = hWbb[ng];
        float lub = lUb[ng], lwub = lWub[ng], lwbb = lWbb[ng];
        #pragma unroll
        for (int mt = 0; mt < 2; ++mt) {
            #pragma unroll
            for (int v = 0; v < 8; ++v) {
                int m = mt * 16 + hi8 + v;
                float o = (ah[mt][v] + hub) * (gh[mt][v] + hwub) + (bh[mt][v] + hwbb)
                        + (al[mt][v] + lub) * (gl[mt][v] + lwub) + (bl[mt][v] + lwbb);
                out[(rowBase + m) * HALF_D + ng] = o;
            }
        }
    }
}

// ---------------------------------------------------------------------------
// Host launcher
// ---------------------------------------------------------------------------
extern "C" void kernel_launch(void* const* d_in, const int* in_sizes, int n_in,
                              void* d_out, int out_size, void* d_ws, size_t ws_size,
                              hipStream_t stream) {
    (void)n_in; (void)out_size; (void)ws_size;

    const int B = in_sizes[0] / HALF_D;   // 32768

    const float* hl   = (const float*)d_in[0];
    const float* hr   = (const float*)d_in[1];
    const float* xw   = (const float*)d_in[2];
    const float* xh   = (const float*)d_in[3];

    // f16 fragment-ordered weights in workspace (total 3.67M f16 = 7.34 MB)
    _Float16* ws = (_Float16*)d_ws;
    size_t off = 0;
    auto conv = [&](int src_idx, int K, int N) -> _Float16* {
        _Float16* dst = ws + off;
        off += (size_t)K * N;
        int total = K * N;
        convert_weight_f16<<<(total + NTHREADS - 1) / NTHREADS, NTHREADS, 0, stream>>>(
            (const float*)d_in[src_idx], dst, K, N);
        return dst;
    };
    _Float16* wqU  = conv(4,  HALF_D, HALF_D);
    _Float16* wqWu = conv(6,  HALF_D, HALF_D);
    _Float16* wqWb = conv(8,  HALF_D, HALF_D);
    _Float16* wkU  = conv(10, HALF_D, HALF_D);
    _Float16* wkWu = conv(12, HALF_D, HALF_D);
    _Float16* wkWb = conv(14, HALF_D, HALF_D);
    _Float16* whU  = conv(18, DIM_D,  HALF_D);
    _Float16* whWu = conv(20, HALF_D, HALF_D);
    _Float16* whWb = conv(22, HALF_D, HALF_D);
    _Float16* wlU  = conv(24, DIM_D,  HALF_D);
    _Float16* wlWu = conv(26, HALF_D, HALF_D);
    _Float16* wlWb = conv(28, HALF_D, HALF_D);

    dim3 grid(B / ROWS);
    child_h_agg_fused<<<grid, NTHREADS, 0, stream>>>(
        hl, hr, xw, xh,
        wqU,  (const float*)d_in[5],
        wqWu, (const float*)d_in[7],
        wqWb, (const float*)d_in[9],
        wkU,  (const float*)d_in[11],
        wkWu, (const float*)d_in[13],
        wkWb, (const float*)d_in[15],
        (const float*)d_in[16], (const float*)d_in[17],
        whU,  (const float*)d_in[19],
        whWu, (const float*)d_in[21],
        whWb, (const float*)d_in[23],
        wlU,  (const float*)d_in[25],
        wlWu, (const float*)d_in[27],
        wlWb, (const float*)d_in[29],
        (float*)d_out);
}
